// predictor_5446018531664
// MI455X (gfx1250) — compile-verified, matrix operand order checked by reading, not run on hardware
//
#include <hip/hip_runtime.h>
#include <hip/hip_bf16.h>

// ---------------- problem constants ----------------
#define BATCH   8192
#define TT      128      // time steps
#define HH      128      // hidden
#define H3      384      // 3*H
#define NSTEPS  30
#define BTILE   16       // batch rows per workgroup
#define NTHREADS 256     // 8 wave32

typedef __attribute__((ext_vector_type(16))) _Float16 v16h;
typedef __attribute__((ext_vector_type(8)))  float    v8f;

// ---------------- LDS layout (bytes), dynamic shared, total 320256 <= 327680 ----------------
// All matrices stored FRAGMENT-MAJOR: each 16x32 f16 fragment = 256 dwords (1KB),
// laid out [q(0..1)][lane(0..31)][4 dwords] so a fragment load = 2x ds_load_b128,
// with consecutive lanes sweeping consecutive LDS banks (no stride-64 conflicts).
#define OFF_W1    0                       // 384x128 f16 = 98304   enc_wih  -> dec_wih
#define OFF_W2    98304                   // 384x128 f16 = 98304   enc_whh  -> dec_whh
#define OFF_UN    196608                  // union region (110592 B)
//   encoder overlay:
#define OFF_PAST  (OFF_UN)                // 16x6x128 f16 = 24576 (row-major, broadcast reads)
#define OFF_CW    (OFF_UN + 24576)        // conv_w f32 128x18 = 9216
//   decoder overlay:
#define OFF_FC1   (OFF_UN)                // 256x128 f16 = 65536
#define OFF_FC2   (OFF_UN + 65536)        // 64x256  f16 = 32768
#define OFF_FC3   (OFF_UN + 98304)        // 16x64   f16 = 2048 (rows>=6 zero-padded)
#define OFF_A1    (OFF_UN + 100352)       // 16x256  f16 = 8192
#define OFF_A2    (OFF_UN + 108544)       // 16x64   f16 = 2048
#define OFF_X     307200                  // 16x128 f16 = 4096 (GRU input / state_past)
#define OFF_H     311296                  // 16x128 f16 = 4096 (hidden state)
#define OFF_BIH   315392                  // 384 f32
#define OFF_BHH   316928                  // 384 f32
#define OFF_FB1   318464                  // 256 f32
#define OFF_FB2   319488                  // 64 f32
#define OFF_CB    319744                  // 128 f32 conv bias
#define SMEM_BYTES 320256

union FragU { v16h h; unsigned u[8]; uint4 q[2]; };

__device__ __forceinline__ unsigned pack2f16(float lo, float hi) {
  _Float16 l = (_Float16)lo, h = (_Float16)hi;
  unsigned short ul = __builtin_bit_cast(unsigned short, l);
  unsigned short uh = __builtin_bit_cast(unsigned short, h);
  return (unsigned)ul | ((unsigned)uh << 16);
}

// Load one swizzled 16x32 fragment #f: two 16B LDS reads per lane.
__device__ __forceinline__ v16h ldfragS(const unsigned* base, int f) {
  const int lane = threadIdx.x & 31;
  const unsigned* p = base + f * 256 + lane * 4;
  FragU fr;
  fr.q[0] = *(const uint4*)p;          // q=0: K offsets {0..7}+8*half
  fr.q[1] = *(const uint4*)(p + 128);  // q=1: K offsets {16..23}+8*half
  return fr.h;
}

// Scatter one f16 element (row r16 in 0..15, absolute K index k) into a swizzled matrix.
__device__ __forceinline__ void st_elem(_Float16* baseH, int r16, int k, _Float16 val) {
  const int kt = k >> 5, koff = k & 31;
  const int lane2 = r16 + (((koff >> 3) & 1) << 4);
  const int dw = kt * 256 + (koff >> 4) * 128 + lane2 * 4 + ((koff & 7) >> 1);
  baseH[dw * 2 + (koff & 1)] = val;
}

// Stage a global f32 matrix [rows][K] into swizzled f16 LDS (rows padded with zeros).
__device__ __forceinline__ void stage_swz(unsigned* dst, const float* __restrict__ src,
                                          int rows, int K, int src_rows) {
  const int ndw = rows * K / 2;
  const int ktc = K >> 5;
  for (int d = threadIdx.x; d < ndw; d += NTHREADS) {
    const int f = d >> 8, w = d & 255;
    const int q = w >> 7, lane = (w >> 2) & 31, vl = w & 3;
    const int r = lane & 15, half = lane >> 4;
    const int koff = vl * 2 + (q << 4) + 8 * half;
    const int nt = f / ktc, kt = f - nt * ktc;
    const int row = nt * 16 + r, k = kt * 32 + koff;
    unsigned u = 0;
    if (row < src_rows) u = pack2f16(src[row * K + k], src[row * K + k + 1]);
    dst[d] = u;
  }
}

__device__ __forceinline__ v8f wmma16(v16h a, v16h b, v8f c) {
  return __builtin_amdgcn_wmma_f32_16x16x32_f16(false, a, false, b, (short)0, c, false, false);
}

__device__ __forceinline__ float sigm(float x) { return 1.f / (1.f + __expf(-x)); }

__global__ __launch_bounds__(NTHREADS)
void gru_predictor_kernel(const float* __restrict__ past,
                          const float* __restrict__ conv_w, const float* __restrict__ conv_b,
                          const float* __restrict__ bn_g, const float* __restrict__ bn_b,
                          const float* __restrict__ bn_m, const float* __restrict__ bn_v,
                          const float* __restrict__ enc_wih, const float* __restrict__ enc_whh,
                          const float* __restrict__ enc_bih, const float* __restrict__ enc_bhh,
                          const float* __restrict__ dec_wih, const float* __restrict__ dec_whh,
                          const float* __restrict__ dec_bih, const float* __restrict__ dec_bhh,
                          const float* __restrict__ fc1_w, const float* __restrict__ fc1_b,
                          const float* __restrict__ fc2_w, const float* __restrict__ fc2_b,
                          const float* __restrict__ fc3_w, const float* __restrict__ fc3_b,
                          float* __restrict__ out)
{
  extern __shared__ char smem[];
  unsigned* W1U  = (unsigned*)(smem + OFF_W1);
  unsigned* W2U  = (unsigned*)(smem + OFF_W2);
  _Float16* Ps   = (_Float16*)(smem + OFF_PAST);
  float*    Wc   = (float*)   (smem + OFF_CW);
  unsigned* FC1U = (unsigned*)(smem + OFF_FC1);
  unsigned* FC2U = (unsigned*)(smem + OFF_FC2);
  unsigned* FC3U = (unsigned*)(smem + OFF_FC3);
  unsigned* A1U  = (unsigned*)(smem + OFF_A1);
  unsigned* A2U  = (unsigned*)(smem + OFF_A2);
  unsigned* XsU  = (unsigned*)(smem + OFF_X);
  unsigned* HsU  = (unsigned*)(smem + OFF_H);
  _Float16* A1H  = (_Float16*)(smem + OFF_A1);
  _Float16* A2H  = (_Float16*)(smem + OFF_A2);
  _Float16* HsH  = (_Float16*)(smem + OFF_H);
  float*    BIH  = (float*)(smem + OFF_BIH);
  float*    BHH  = (float*)(smem + OFF_BHH);
  float*    FB1  = (float*)(smem + OFF_FB1);
  float*    FB2  = (float*)(smem + OFF_FB2);
  float*    Cb   = (float*)(smem + OFF_CB);

  const int tid  = threadIdx.x;
  const int wv   = tid >> 5;        // wave id 0..7
  const int lane = tid & 31;
  const int ln   = lane & 15;       // N-lane / M-row within fragment
  const int half = lane >> 4;
  const int jw   = wv * 16;         // this wave's hidden-column block
  const int b0   = blockIdx.x * BTILE;

  // ---------------- phase 0: stage encoder weights (swizzled) + past tile ----------------
  stage_swz(W1U, enc_wih, H3, HH, H3);
  stage_swz(W2U, enc_whh, H3, HH, H3);
  for (int i = tid; i < BTILE * 6 * TT; i += NTHREADS) Ps[i] = (_Float16)past[b0 * 6 * TT + i];
  for (int i = tid; i < HH * 18; i += NTHREADS) Wc[i] = conv_w[i];
  for (int i = tid; i < HH;      i += NTHREADS) Cb[i] = conv_b[i];
  for (int i = tid; i < H3;      i += NTHREADS) { BIH[i] = enc_bih[i]; BHH[i] = enc_bhh[i]; }
  for (int i = tid; i < BTILE * HH / 2; i += NTHREADS) HsU[i] = 0u;
  __syncthreads();

  const float bnscale = bn_g[0] * rsqrtf(bn_v[0] + 1e-5f);
  const float bnshift = bn_b[0] - bn_m[0] * bnscale;

  float bihr = BIH[jw + ln], bihz = BIH[128 + jw + ln], bihn = BIH[256 + jw + ln];
  float bhhr = BHH[jw + ln], bhhz = BHH[128 + jw + ln], bhhn = BHH[256 + jw + ln];

  float hreg[8];
#pragma unroll
  for (int r = 0; r < 8; ++r) hreg[r] = 0.f;

  const int cm = tid >> 4;          // conv: batch row 0..15
  const int cc = (tid & 15) * 8;    // conv: channel base (8 contiguous channels)
  // destination of this thread's 8 conv outputs in swizzled Xs: one contiguous uint4
  const int ckt = cc >> 5, ckoff = cc & 31;
  unsigned* cdst = XsU + ckt * 256 + (ckoff >> 4) * 128 + (cm + (((ckoff >> 3) & 1) << 4)) * 4;

  // ================= encoder: 128 recurrent steps =================
  for (int t = 0; t < TT; ++t) {
    // ---- conv1d(k=3,pad=1) + relu + BN for time t -> Xs (swizzled f16) ----
    float patch[18];
#pragma unroll
    for (int i = 0; i < 6; ++i)
#pragma unroll
      for (int k = 0; k < 3; ++k) {
        int ts = t + k - 1;
        patch[i * 3 + k] = (ts < 0 || ts >= TT) ? 0.f : (float)Ps[(cm * 6 + i) * TT + ts];
      }
    float cv[8];
#pragma unroll
    for (int c8 = 0; c8 < 8; ++c8) {
      const int c = cc + c8;
      float acc = Cb[c];
#pragma unroll
      for (int q = 0; q < 18; ++q) acc = fmaf(patch[q], Wc[c * 18 + q], acc);
      cv[c8] = fmaxf(acc, 0.f) * bnscale + bnshift;
    }
    uint4 pk;
    pk.x = pack2f16(cv[0], cv[1]); pk.y = pack2f16(cv[2], cv[3]);
    pk.z = pack2f16(cv[4], cv[5]); pk.w = pack2f16(cv[6], cv[7]);
    *(uint4*)cdst = pk;
    __syncthreads();

    // ---- preload A fragments (x_t and h_{t-1}) before anyone rewrites them ----
    v16h ax[4], ah[4];
#pragma unroll
    for (int k = 0; k < 4; ++k) { ax[k] = ldfragS(XsU, k); ah[k] = ldfragS(HsU, k); }
    __syncthreads();

    // ---- gi = x@Wih^T, gh = h@Whh^T : 6 accumulator tiles, 24 WMMAs ----
    v8f air = {}, aiz = {}, ain = {}, ahr = {}, ahz = {}, ahn = {};
#pragma unroll
    for (int k = 0; k < 4; ++k) {
      air = wmma16(ax[k], ldfragS(W1U, (0 * 8 + wv) * 4 + k), air);
      aiz = wmma16(ax[k], ldfragS(W1U, (1 * 8 + wv) * 4 + k), aiz);
      ain = wmma16(ax[k], ldfragS(W1U, (2 * 8 + wv) * 4 + k), ain);
      ahr = wmma16(ah[k], ldfragS(W2U, (0 * 8 + wv) * 4 + k), ahr);
      ahz = wmma16(ah[k], ldfragS(W2U, (1 * 8 + wv) * 4 + k), ahz);
      ahn = wmma16(ah[k], ldfragS(W2U, (2 * 8 + wv) * 4 + k), ahn);
    }
    // ---- gates + state update (wave-local columns jw..jw+15) ----
#pragma unroll
    for (int r = 0; r < 8; ++r) {
      float rg = sigm(air[r] + bihr + ahr[r] + bhhr);
      float zg = sigm(aiz[r] + bihz + ahz[r] + bhhz);
      float ng = tanhf(ain[r] + bihn + rg * (ahn[r] + bhhn));
      float hn = (1.f - zg) * ng + zg * hreg[r];
      hreg[r] = hn;
      st_elem(HsH, r + 8 * half, jw + ln, (_Float16)hn);
    }
  }
  __syncthreads();

  // ================= phase switch: decoder weights into LDS =================
  for (int i = tid; i < BTILE * HH / 2; i += NTHREADS) { XsU[i] = HsU[i]; HsU[i] = 0u; }
  stage_swz(W1U, dec_wih, H3, HH, H3);
  stage_swz(W2U, dec_whh, H3, HH, H3);
  stage_swz(FC1U, fc1_w, 256, HH, 256);
  stage_swz(FC2U, fc2_w, 64, 256, 64);
  stage_swz(FC3U, fc3_w, 16, 64, 6);          // rows 6..15 zero
  for (int i = tid; i < H3;  i += NTHREADS) { BIH[i] = dec_bih[i]; BHH[i] = dec_bhh[i]; }
  for (int i = tid; i < 256; i += NTHREADS) FB1[i] = fc1_b[i];
  for (int i = tid; i < 64;  i += NTHREADS) FB2[i] = fc2_b[i];
  __syncthreads();

  bihr = BIH[jw + ln]; bihz = BIH[128 + jw + ln]; bihn = BIH[256 + jw + ln];
  bhhr = BHH[jw + ln]; bhhz = BHH[128 + jw + ln]; bhhn = BHH[256 + jw + ln];
#pragma unroll
  for (int r = 0; r < 8; ++r) hreg[r] = 0.f;

  // running `present` lives in wave0's FC3 accumulator (C chains across steps)
  v8f pres = {};
  float fb3 = 0.f;
  if (wv == 0) {
    fb3 = (ln < 6) ? fc3_b[ln] : 0.f;
#pragma unroll
    for (int r = 0; r < 8; ++r)
      pres[r] = (ln < 6) ? past[((b0 + r + 8 * half) * 6 + ln) * TT + (TT - 1)] : 0.f;
  }

  v16h axd[4];   // state_past A-fragments (constant; only used at s==0)
#pragma unroll
  for (int k = 0; k < 4; ++k) axd[k] = ldfragS(XsU, k);

  // ================= decoder: 30 steps of GRU + MLP =================
  for (int s = 0; s < NSTEPS; ++s) {
    v16h ah[4];
#pragma unroll
    for (int k = 0; k < 4; ++k) ah[k] = ldfragS(HsU, k);
    __syncthreads();

    v8f air = {}, aiz = {}, ain = {}, ahr = {}, ahz = {}, ahn = {};
#pragma unroll
    for (int k = 0; k < 4; ++k) {
      ahr = wmma16(ah[k], ldfragS(W2U, (0 * 8 + wv) * 4 + k), ahr);
      ahz = wmma16(ah[k], ldfragS(W2U, (1 * 8 + wv) * 4 + k), ahz);
      ahn = wmma16(ah[k], ldfragS(W2U, (2 * 8 + wv) * 4 + k), ahn);
    }
    if (s == 0) {   // x nonzero only at step 0 (x = state_past); else gi = bias
#pragma unroll
      for (int k = 0; k < 4; ++k) {
        air = wmma16(axd[k], ldfragS(W1U, (0 * 8 + wv) * 4 + k), air);
        aiz = wmma16(axd[k], ldfragS(W1U, (1 * 8 + wv) * 4 + k), aiz);
        ain = wmma16(axd[k], ldfragS(W1U, (2 * 8 + wv) * 4 + k), ain);
      }
    }
#pragma unroll
    for (int r = 0; r < 8; ++r) {
      float rg = sigm(air[r] + bihr + ahr[r] + bhhr);
      float zg = sigm(aiz[r] + bihz + ahz[r] + bhhz);
      float ng = tanhf(ain[r] + bihn + rg * (ahn[r] + bhhn));
      float hn = (1.f - zg) * ng + zg * hreg[r];
      hreg[r] = hn;
      st_elem(HsH, r + 8 * half, jw + ln, (_Float16)hn);
    }
    __syncthreads();

    // ---- fc1: (16x128)@(128x256) + relu, 2 N-tiles per wave ----
    {
      v16h a[4];
#pragma unroll
      for (int k = 0; k < 4; ++k) a[k] = ldfragS(HsU, k);
      v8f c0 = {}, c1 = {};
      const int n0 = wv * 32;
#pragma unroll
      for (int k = 0; k < 4; ++k) {
        c0 = wmma16(a[k], ldfragS(FC1U, (wv * 2 + 0) * 4 + k), c0);
        c1 = wmma16(a[k], ldfragS(FC1U, (wv * 2 + 1) * 4 + k), c1);
      }
      const float bb0 = FB1[n0 + ln], bb1 = FB1[n0 + 16 + ln];
#pragma unroll
      for (int r = 0; r < 8; ++r) {
        const int M = r + 8 * half;
        st_elem(A1H, M, n0 + ln,      (_Float16)fmaxf(c0[r] + bb0, 0.f));
        st_elem(A1H, M, n0 + 16 + ln, (_Float16)fmaxf(c1[r] + bb1, 0.f));
      }
    }
    __syncthreads();

    // ---- fc2: (16x256)@(256x64), waves 0..3 ----
    if (wv < 4) {
      v8f c = {};
      const int n0 = wv * 16;
#pragma unroll
      for (int k = 0; k < 8; ++k) {
        v16h a = ldfragS(A1U, k);
        c = wmma16(a, ldfragS(FC2U, wv * 8 + k), c);
      }
      const float bb = FB2[n0 + ln];
#pragma unroll
      for (int r = 0; r < 8; ++r)
        st_elem(A2H, r + 8 * half, n0 + ln, (_Float16)(c[r] + bb));
    }
    __syncthreads();

    // ---- fc3: (16x64)@(64x6 padded to 16), wave 0; C = running present ----
    if (wv == 0) {
#pragma unroll
      for (int k = 0; k < 2; ++k) {
        v16h a = ldfragS(A2U, k);
        pres = wmma16(a, ldfragS(FC3U, k), pres);
      }
#pragma unroll
      for (int r = 0; r < 8; ++r) {
        pres[r] += fb3;
        if (ln < 6)
          out[((b0 + r + 8 * half) * 6 + ln) * NSTEPS + s] = pres[r];
      }
    }
    __syncthreads();
  }
}

extern "C" void kernel_launch(void* const* d_in, const int* in_sizes, int n_in,
                              void* d_out, int out_size, void* d_ws, size_t ws_size,
                              hipStream_t stream) {
  (void)in_sizes; (void)n_in; (void)out_size; (void)d_ws; (void)ws_size;
  const float* past    = (const float*)d_in[0];
  const float* conv_w  = (const float*)d_in[1];
  const float* conv_b  = (const float*)d_in[2];
  const float* bn_g    = (const float*)d_in[3];
  const float* bn_b    = (const float*)d_in[4];
  const float* bn_m    = (const float*)d_in[5];
  const float* bn_v    = (const float*)d_in[6];
  const float* enc_wih = (const float*)d_in[7];
  const float* enc_whh = (const float*)d_in[8];
  const float* enc_bih = (const float*)d_in[9];
  const float* enc_bhh = (const float*)d_in[10];
  const float* dec_wih = (const float*)d_in[11];
  const float* dec_whh = (const float*)d_in[12];
  const float* dec_bih = (const float*)d_in[13];
  const float* dec_bhh = (const float*)d_in[14];
  const float* fc1_w   = (const float*)d_in[15];
  const float* fc1_b   = (const float*)d_in[16];
  const float* fc2_w   = (const float*)d_in[17];
  const float* fc2_b   = (const float*)d_in[18];
  const float* fc3_w   = (const float*)d_in[19];
  const float* fc3_b   = (const float*)d_in[20];
  float* out = (float*)d_out;

  // allow >64KB dynamic LDS (capture-safe; not a stream op)
  (void)hipFuncSetAttribute((const void*)gru_predictor_kernel,
                            hipFuncAttributeMaxDynamicSharedMemorySize, SMEM_BYTES);

  dim3 grid(BATCH / BTILE), block(NTHREADS);
  gru_predictor_kernel<<<grid, block, SMEM_BYTES, stream>>>(
      past, conv_w, conv_b, bn_g, bn_b, bn_m, bn_v,
      enc_wih, enc_whh, enc_bih, enc_bhh,
      dec_wih, dec_whh, dec_bih, dec_bhh,
      fc1_w, fc1_b, fc2_w, fc2_b, fc3_w, fc3_b, out);
}
